// Model_78254304133638
// MI455X (gfx1250) — compile-verified, matrix operand order checked by reading, not run on hardware
//
#include <hip/hip_runtime.h>
#include <hip/hip_bf16.h>

// ---------------------------------------------------------------------------
// Transformer encoder-decoder forward for MI455X (gfx1250), bf16 WMMA GEMMs
// with f32 accumulation. All matrix math runs through v_wmma_f32_16x16x32_bf16.
// Double-buffered LDS staging overlaps HBM loads with WMMA compute.
// ---------------------------------------------------------------------------

typedef __attribute__((ext_vector_type(16))) __bf16 bf16x16;
typedef __attribute__((ext_vector_type(8)))  __bf16 bf16x8;
typedef __attribute__((ext_vector_type(4)))  __bf16 bf16x4;
typedef __attribute__((ext_vector_type(8)))  float  f32x8;
typedef __attribute__((ext_vector_type(4)))  float  f32x4;

#define TM 64
#define TN 128
#define TK 64

// ---------------------------------------------------------------------------
// Generic GEMM: C[M,N] = alpha * A[M,K] (x) op(B) + bias, optional ReLU,
// optional transposed store (C[n*ldct+m]).  Batched via grid.z with the batch
// index decomposed z = zb*batchH + zh (per-(b,h) head GEMMs with non-uniform
// strides).  A,B are f32 in memory, converted to bf16 while staging to LDS.
//
// WMMA fragment layouts per CDNA5 ISA 7.12.2 (wave32):
//   A 16x32 bf16: lane m=l&15, K-half = (l>>4)*8; VGPR0-3 = K[kh..kh+7],
//                 VGPR4-7 = K[16+kh..16+kh+7]  -> two ds_load_b128 per frag.
//   B 32x16 bf16: symmetric, lane holds column n=l&15.
//   C 16x16 f32 : lane col n=l&15, VGPR r holds row r + ((l>=16)?8:0).
// ---------------------------------------------------------------------------
__launch_bounds__(256)
__global__ void gemm_wmma_bf16(
    const float* __restrict__ A, const float* __restrict__ Bm,
    const float* __restrict__ bias, float* __restrict__ C,
    int M, int N, int K, int lda, int ldb, int ldc,
    int transB, int relu, float alpha, int storeT, int ldct,
    int batchH,
    long long sAb, long long sAh, long long sBb, long long sBh,
    long long sCb, long long sCh)
{
    __shared__ __bf16 As[2][TM][TK];   // 16 KB
    __shared__ __bf16 Bs[2][TN][TK];   // 32 KB

    const int z  = blockIdx.z;
    const int zb = z / batchH, zh = z % batchH;
    A  += (size_t)zb * sAb + (size_t)zh * sAh;
    Bm += (size_t)zb * sBb + (size_t)zh * sBh;
    C  += (size_t)zb * sCb + (size_t)zh * sCh;

    const int tid  = threadIdx.x;
    const int lane = tid & 31;
    const int m0 = blockIdx.y * TM;
    const int n0 = blockIdx.x * TN;

    const int wave = tid >> 5;            // 8 waves: 2 (M) x 4 (N)
    const int wm = (wave >> 2) * 32;
    const int wn = (wave & 3) * 32;

    f32x8 acc[2][2] = {};

    const int lm = lane & 15;             // row/col within fragment
    const int kh = (lane >> 4) * 8;       // K-half offset (elements)

    // ---- staging helpers (load global -> regs; convert -> LDS buffer p) --
    auto loadA = [&](int k0, f32x4* av) {
#pragma unroll
        for (int e = 0; e < 4; ++e) {
            int i = tid + e * 256;
            int r = i >> 4, c4 = (i & 15) * 4;          // 16 chunks per row
            av[e] = *(const f32x4*)(A + (size_t)(m0 + r) * lda + k0 + c4);
        }
    };
    auto storeA = [&](int p, const f32x4* av) {
#pragma unroll
        for (int e = 0; e < 4; ++e) {
            int i = tid + e * 256;
            int r = i >> 4, c4 = (i & 15) * 4;
            bf16x4 o;
            o[0] = (__bf16)av[e][0]; o[1] = (__bf16)av[e][1];
            o[2] = (__bf16)av[e][2]; o[3] = (__bf16)av[e][3];
            *(bf16x4*)&As[p][r][c4] = o;
        }
    };
    auto loadB = [&](int k0, f32x4* bv) {
        if (!transB) {
            // [K,N] row-major: 4x4 (k x n) units, 2 per thread
#pragma unroll
            for (int e = 0; e < 2; ++e) {
                int u = tid + e * 256;                   // 512 units
                int ku = (u >> 5) * 4, nu = (u & 31) * 4;
#pragma unroll
                for (int r = 0; r < 4; ++r) {
                    f32x4 v = {0.f, 0.f, 0.f, 0.f};
                    if (n0 + nu < N)
                        v = *(const f32x4*)(Bm + (size_t)(k0 + ku + r) * ldb + (n0 + nu));
                    bv[e * 4 + r] = v;
                }
            }
        } else {
            // [N,K] row-major: float4 along k
#pragma unroll
            for (int e = 0; e < 8; ++e) {
                int i = tid + e * 256;
                int nn = i >> 4, k4 = (i & 15) * 4;      // 16 chunks per row
                f32x4 v = {0.f, 0.f, 0.f, 0.f};
                if (n0 + nn < N)
                    v = *(const f32x4*)(Bm + (size_t)(n0 + nn) * ldb + k0 + k4);
                bv[e] = v;
            }
        }
    };
    auto storeB = [&](int p, const f32x4* bv) {
        if (!transB) {
#pragma unroll
            for (int e = 0; e < 2; ++e) {
                int u = tid + e * 256;
                int ku = (u >> 5) * 4, nu = (u & 31) * 4;
#pragma unroll
                for (int j = 0; j < 4; ++j) {
                    bf16x4 o;                             // pack 4 k's for col nu+j
                    o[0] = (__bf16)bv[e * 4 + 0][j]; o[1] = (__bf16)bv[e * 4 + 1][j];
                    o[2] = (__bf16)bv[e * 4 + 2][j]; o[3] = (__bf16)bv[e * 4 + 3][j];
                    *(bf16x4*)&Bs[p][nu + j][ku] = o;
                }
            }
        } else {
#pragma unroll
            for (int e = 0; e < 8; ++e) {
                int i = tid + e * 256;
                int nn = i >> 4, k4 = (i & 15) * 4;
                bf16x4 o;
                o[0] = (__bf16)bv[e][0]; o[1] = (__bf16)bv[e][1];
                o[2] = (__bf16)bv[e][2]; o[3] = (__bf16)bv[e][3];
                *(bf16x4*)&Bs[p][nn][k4] = o;
            }
        }
    };

    // ---- prologue: stage tile 0 ------------------------------------------
    {
        f32x4 av[4], bv[8];
        loadA(0, av); loadB(0, bv);
        storeA(0, av); storeB(0, bv);
    }
    __syncthreads();

    int cur = 0;
    for (int k0 = 0; k0 < K; k0 += TK) {
        const int kn = k0 + TK;
        const bool more = kn < K;

        // issue next tile's global loads before the WMMA block (latency hide)
        f32x4 av[4], bv[8];
        if (more) {
            loadA(kn, av);
            loadB(kn, bv);
            if (kn + TK < K) {          // warm L2 two tiles ahead
                __builtin_prefetch(A + (size_t)(m0 + (tid >> 4)) * lda + kn + TK, 0, 1);
                __builtin_prefetch(Bm + (transB ? (size_t)(n0 + (tid >> 4)) * ldb + kn + TK
                                                : (size_t)(kn + TK + (tid >> 5)) * ldb + n0), 0, 1);
            }
        }

        // ---- 2 K-steps of 32: fragment loads + 4 wmma each ---------------
#pragma unroll
        for (int ks = 0; ks < TK; ks += 32) {
            bf16x16 afrag[2], bfrag[2];
#pragma unroll
            for (int i = 0; i < 2; ++i) {
                const __bf16* rp = &As[cur][wm + i * 16 + lm][ks];
                union { bf16x16 v; bf16x8 h[2]; } u;
                u.h[0] = *(const bf16x8*)(rp + kh);
                u.h[1] = *(const bf16x8*)(rp + 16 + kh);
                afrag[i] = u.v;
            }
#pragma unroll
            for (int j = 0; j < 2; ++j) {
                const __bf16* cp = &Bs[cur][wn + j * 16 + lm][ks];
                union { bf16x16 v; bf16x8 h[2]; } u;
                u.h[0] = *(const bf16x8*)(cp + kh);
                u.h[1] = *(const bf16x8*)(cp + 16 + kh);
                bfrag[j] = u.v;
            }
#pragma unroll
            for (int i = 0; i < 2; ++i)
#pragma unroll
                for (int j = 0; j < 2; ++j)
                    acc[i][j] = __builtin_amdgcn_wmma_f32_16x16x32_bf16(
                        false, afrag[i], false, bfrag[j],
                        (short)0, acc[i][j], false, false);
        }

        // convert + store next tile into the alternate buffer
        if (more) {
            storeA(1 - cur, av);
            storeB(1 - cur, bv);
        }
        __syncthreads();
        cur ^= 1;
    }

    // ---- epilogue: alpha, bias, relu, (optionally transposed) store ------
    const int rbase = (lane >> 4) * 8;
    const int cl = lane & 15;
#pragma unroll
    for (int i = 0; i < 2; ++i) {
#pragma unroll
        for (int j = 0; j < 2; ++j) {
            int col = n0 + wn + j * 16 + cl;
            if (col >= N) continue;
            float bv = bias ? bias[col] : 0.f;
#pragma unroll
            for (int r = 0; r < 8; ++r) {
                int row = m0 + wm + i * 16 + r + rbase;
                float v = acc[i][j][r] * alpha + bv;
                if (relu) v = fmaxf(v, 0.f);
                if (storeT) C[(size_t)col * ldct + row] = v;
                else        C[(size_t)row * ldc  + col] = v;
            }
        }
    }
}

// ---------------------------------------------------------------------------
// Row softmax over n=1024 contiguous elements, one block per row.
// ---------------------------------------------------------------------------
__launch_bounds__(256)
__global__ void softmax_rows(float* __restrict__ S, int n)
{
    __shared__ float red[256];
    float* row = S + (size_t)blockIdx.x * n;
    int tid = threadIdx.x;

    float m = -3.4e38f;
    for (int i = tid; i < n; i += 256) m = fmaxf(m, row[i]);
    red[tid] = m; __syncthreads();
    for (int s = 128; s > 0; s >>= 1) {
        if (tid < s) red[tid] = fmaxf(red[tid], red[tid + s]);
        __syncthreads();
    }
    m = red[0]; __syncthreads();

    float sum = 0.f;
    for (int i = tid; i < n; i += 256) {
        float e = __expf(row[i] - m);
        row[i] = e; sum += e;
    }
    red[tid] = sum; __syncthreads();
    for (int s = 128; s > 0; s >>= 1) {
        if (tid < s) red[tid] += red[tid + s];
        __syncthreads();
    }
    float inv = 1.f / red[0];
    for (int i = tid; i < n; i += 256) row[i] *= inv;
}

// ---------------------------------------------------------------------------
// X = LayerNorm(T + X) * g + b   (row = token, D = 512, in-place on X)
// ---------------------------------------------------------------------------
__launch_bounds__(256)
__global__ void ln_residual(const float* __restrict__ T, float* __restrict__ X,
                            const float* __restrict__ g, const float* __restrict__ b)
{
    __shared__ float red[256];
    size_t base = (size_t)blockIdx.x * 512;
    int tid = threadIdx.x;

    float v0 = T[base + tid]       + X[base + tid];
    float v1 = T[base + tid + 256] + X[base + tid + 256];
    red[tid] = v0 + v1; __syncthreads();
    for (int s = 128; s > 0; s >>= 1) {
        if (tid < s) red[tid] += red[tid + s];
        __syncthreads();
    }
    float mu = red[0] * (1.f / 512.f); __syncthreads();

    float d0 = v0 - mu, d1 = v1 - mu;
    red[tid] = d0 * d0 + d1 * d1; __syncthreads();
    for (int s = 128; s > 0; s >>= 1) {
        if (tid < s) red[tid] += red[tid + s];
        __syncthreads();
    }
    float inv = rsqrtf(red[0] * (1.f / 512.f) + 1e-5f);

    X[base + tid]       = d0 * inv * g[tid]       + b[tid];
    X[base + tid + 256] = d1 * inv * g[tid + 256] + b[tid + 256];
}

// ---------------------------------------------------------------------------
// Embedding gather: out[t,:] = emb[idx[t],:]  (D = 512, one token per block)
// ---------------------------------------------------------------------------
__launch_bounds__(256)
__global__ void gather_emb(const int* __restrict__ idx, const float* __restrict__ emb,
                           float* __restrict__ out)
{
    int t = blockIdx.x;
    size_t src = (size_t)idx[t] * 512;
    size_t dst = (size_t)t * 512;
    out[dst + threadIdx.x]       = emb[src + threadIdx.x];
    out[dst + threadIdx.x + 256] = emb[src + threadIdx.x + 256];
}

// ---------------------------------------------------------------------------
// Host-side orchestration
// ---------------------------------------------------------------------------
namespace {

struct AttnP { const float *Wq,*bq,*Wk,*bk,*Wv,*bv,*Wo,*bo,*g,*b; };
struct FfnP  { const float *W1,*b1,*W2,*b2,*g,*b; };

inline void gemm(hipStream_t s, const float* A, const float* B, const float* bias,
                 float* C, int M, int N, int K, int lda, int ldb, int ldc,
                 int transB, int relu, float alpha,
                 int storeT = 0, int ldct = 0, int batch = 1, int batchH = 1,
                 long long sAb = 0, long long sAh = 0,
                 long long sBb = 0, long long sBh = 0,
                 long long sCb = 0, long long sCh = 0)
{
    dim3 grid((N + TN - 1) / TN, (M + TM - 1) / TM, batch);
    gemm_wmma_bf16<<<grid, 256, 0, s>>>(A, B, bias, C, M, N, K, lda, ldb, ldc,
                                        transB, relu, alpha, storeT, ldct,
                                        batchH, sAb, sAh, sBb, sBh, sCb, sCh);
}

// xq: [2048, 512] (updated in place), xkv: [2048, 512]
void run_mha(hipStream_t s, float* xq, const float* xkv, const AttnP& p,
             float* Q, float* Kb, float* Vb, float* Ctx, float* T, float* Sc)
{
    const int M = 2048;
    gemm(s, xq,  p.Wq, p.bq, Q,  M, 512, 512, 512, 512, 512, 0, 0, 1.f);
    gemm(s, xkv, p.Wk, p.bk, Kb, M, 512, 512, 512, 512, 512, 0, 0, 1.f);
    gemm(s, xkv, p.Wv, p.bv, Vb, M, 512, 512, 512, 512, 512, 0, 0, 1.f);
    // scores[b,h] = (Q_h K_h^T)/8 ;  16 batched (b,h) GEMMs, M=N=1024, K=64
    gemm(s, Q, Kb, nullptr, Sc, 1024, 1024, 64, 512, 512, 1024,
         /*transB*/1, 0, 0.125f, 0, 0, /*batch*/16, /*batchH*/8,
         1024LL * 512, 64, 1024LL * 512, 64,
         8LL * 1024 * 1024, 1024LL * 1024);
    softmax_rows<<<16 * 1024, 256, 0, s>>>(Sc, 1024);
    // ctx[b,:,h] = attn[b,h] @ V_h ;  M=1024, N=64, K=1024
    gemm(s, Sc, Vb, nullptr, Ctx, 1024, 64, 1024, 1024, 512, 512,
         0, 0, 1.f, 0, 0, 16, 8,
         8LL * 1024 * 1024, 1024LL * 1024,
         1024LL * 512, 64, 1024LL * 512, 64);
    gemm(s, Ctx, p.Wo, p.bo, T, M, 512, 512, 512, 512, 512, 0, 0, 1.f);
    ln_residual<<<2048, 256, 0, s>>>(T, xq, p.g, p.b);
}

void run_ffn(hipStream_t s, float* x, const FfnP& p, float* Hb, float* T)
{
    gemm(s, x,  p.W1, p.b1, Hb, 2048, 2048, 512,  512,  2048, 2048, 0, /*relu*/1, 1.f);
    gemm(s, Hb, p.W2, p.b2, T,  2048, 512,  2048, 2048, 512,  512,  0, 0, 1.f);
    ln_residual<<<2048, 256, 0, s>>>(T, x, p.g, p.b);
}

} // namespace

extern "C" void kernel_launch(void* const* d_in, const int* in_sizes, int n_in,
                              void* d_out, int out_size, void* d_ws, size_t ws_size,
                              hipStream_t stream)
{
    (void)in_sizes; (void)n_in; (void)out_size; (void)ws_size;
    auto F = [&](int i) { return (const float*)d_in[i]; };

    // setup_inputs() dict insertion order, nested dicts flattened in order:
    // 0 input_indices  1 output_indices  2 emb_enc  3 emb_dec  4 out_W  5 out_b
    // 6..15  enc.attn  (Wq,bq,Wk,bk,Wv,bv,Wo,bo,ln_g,ln_b)
    // 16..21 enc.ffn   (W1,b1,W2,b2,ln_g,ln_b)
    // 22..31 dec.self  32..41 dec.cross  42..47 dec.ffn
    const int* in_idx  = (const int*)d_in[0];
    const int* out_idx = (const int*)d_in[1];
    const float* emb_enc = F(2);
    const float* emb_dec = F(3);
    const float* out_W   = F(4);
    const float* out_b   = F(5);

    float* ws = (float*)d_ws;
    const size_t ACT = 2ull * 1024 * 512;          // 1,048,576 f32
    float* X   = ws;                                // encoder activations
    float* Y   = ws + ACT;                          // decoder activations
    float* Q   = ws + 2 * ACT;
    float* Kb  = ws + 3 * ACT;
    float* Vb  = ws + 4 * ACT;
    float* Ctx = ws + 5 * ACT;
    float* T   = ws + 6 * ACT;                      // projection tmp
    float* Hb  = ws + 7 * ACT;                      // FFN hidden (4*ACT)
    float* Sc  = ws + 11 * ACT;                     // scores (16*ACT)

    auto attnP = [&](int base, int l) {
        AttnP p;
        p.Wq = F(base + 0) + (size_t)l * 512 * 512; p.bq = F(base + 1) + (size_t)l * 512;
        p.Wk = F(base + 2) + (size_t)l * 512 * 512; p.bk = F(base + 3) + (size_t)l * 512;
        p.Wv = F(base + 4) + (size_t)l * 512 * 512; p.bv = F(base + 5) + (size_t)l * 512;
        p.Wo = F(base + 6) + (size_t)l * 512 * 512; p.bo = F(base + 7) + (size_t)l * 512;
        p.g  = F(base + 8) + (size_t)l * 512;       p.b  = F(base + 9) + (size_t)l * 512;
        return p;
    };
    auto ffnP = [&](int base, int l) {
        FfnP p;
        p.W1 = F(base + 0) + (size_t)l * 512 * 2048; p.b1 = F(base + 1) + (size_t)l * 2048;
        p.W2 = F(base + 2) + (size_t)l * 2048 * 512; p.b2 = F(base + 3) + (size_t)l * 512;
        p.g  = F(base + 4) + (size_t)l * 512;        p.b  = F(base + 5) + (size_t)l * 512;
        return p;
    };

    // -------- encoder --------
    gather_emb<<<2048, 256, 0, stream>>>(in_idx, emb_enc, X);
    for (int i = 0; i < 4; ++i) {
        run_mha(stream, X, X, attnP(6, i), Q, Kb, Vb, Ctx, T, Sc);
        run_ffn(stream, X, ffnP(16, i), Hb, T);
    }

    // -------- decoder --------
    gather_emb<<<2048, 256, 0, stream>>>(out_idx, emb_dec, Y);
    for (int i = 0; i < 4; ++i) {
        run_mha(stream, Y, Y, attnP(22, i), Q, Kb, Vb, Ctx, T, Sc);   // self
        run_mha(stream, Y, X, attnP(32, i), Q, Kb, Vb, Ctx, T, Sc);   // cross
        run_ffn(stream, Y, ffnP(42, i), Hb, T);
    }

    // -------- logits: y @ out_W + out_b, stored transposed as [B, V, S] ----
    gemm(stream, Y, out_W, out_b, (float*)d_out,
         1024, 32000, 512, 512, 32000, 0, /*transB*/0, 0, 1.f,
         /*storeT*/1, /*ldct*/1024, /*batch*/2, 1,
         1024LL * 512, 0, 0, 0, 32000LL * 1024, 0);
}
